// DINAttention_83545703842342
// MI455X (gfx1250) — matrix operand order repaired and checked by reading, not compiled, not run on hardware
//
#include <hip/hip_runtime.h>

typedef __bf16 bf16;
typedef __attribute__((ext_vector_type(16))) __bf16 v16bf;
typedef __attribute__((ext_vector_type(8)))  __bf16 v8bf;
typedef __attribute__((ext_vector_type(4)))  __bf16 v4bf;
typedef __attribute__((ext_vector_type(8)))  float  v8f;
typedef __attribute__((ext_vector_type(4)))  float  v4f;

#define BATCH 2048
#define SEQ   200
#define DIM   128
#define NT0   5     // 80 / 16 output tiles, layer 0
#define KT0   4     // 128 / 32 k-steps, layer 0 (after feat decomposition)
#define NT1   3     // 48 / 16 output tiles, layer 1 (40 padded to 48)
#define KT1   3     // 96 / 32 k-steps, layer 1 (80 padded to 96)
#define HSTRIDE  136   // halves; 272B row stride -> conflict-free A-tile reads
#define H0STRIDE 104   // halves; 208B row stride -> conflict-free A-tile reads

// ws layout (bf16 element offsets)
#define WS_W0AC 0
#define WS_W0D  20480
#define WS_W1   40960
#define WS_TOTAL 50176

// Pre-swizzle weights into WMMA B-tile lane order (bf16).
// B-tile element [lane][h] = W[kt*32 + (lane>=16?16:0) + h][nt*16 + (lane&15)]
__global__ __launch_bounds__(256) void din_prep(const float* __restrict__ W0,
                                                const float* __restrict__ W1,
                                                bf16* __restrict__ ws)
{
  int idx = blockIdx.x * 256 + threadIdx.x;
  if (idx >= WS_TOTAL) return;
  int e     = idx & 1023;
  int lane  = e >> 4;
  int h     = e & 15;
  int khalf = (lane >> 4) << 4;   // 0 or 16
  int mrow  = lane & 15;
  if (idx < WS_W0D) {
    int tile = idx >> 10;
    int kt = tile / NT0, nt = tile % NT0;
    int k = kt*32 + khalf + h;
    int n = nt*16 + mrow;
    // W0ac = W0[hist rows] + W0[diff rows]
    ws[idx] = (bf16)(W0[k*80 + n] + W0[(256 + k)*80 + n]);
  } else if (idx < WS_W1) {
    int tile = (idx - WS_W0D) >> 10;
    int kt = tile / NT0, nt = tile % NT0;
    int k = kt*32 + khalf + h;
    int n = nt*16 + mrow;
    // W0d = W0[prod rows]
    ws[idx] = (bf16)(W0[(384 + k)*80 + n]);
  } else {
    int tile = (idx - WS_W1) >> 10;
    int kt = tile / NT1, nt = tile % NT1;
    int k = kt*32 + khalf + h;
    int n = nt*16 + mrow;
    float v = (k < 80 && n < 40) ? W1[k*40 + n] : 0.f;  // zero-pad K->96, N->48
    ws[idx] = (bf16)v;
  }
}

static __device__ __forceinline__ v8f wmma_bf16(v16bf a, v16bf b, v8f c) {
  return __builtin_amdgcn_wmma_f32_16x16x32_bf16(false, a, false, b,
                                                 (short)0, c, false, false);
}

__global__ __launch_bounds__(256) void din_main(
    const float* __restrict__ hist, const float* __restrict__ tgt,
    const int*   __restrict__ mask, const float* __restrict__ W0,
    const float* __restrict__ b0,   const float* __restrict__ b1,
    const float* __restrict__ W2,   const float* __restrict__ b2,
    const bf16*  __restrict__ ws,
    float* __restrict__ out_ui,     float* __restrict__ out_attn)
{
  __shared__ __align__(16) bf16  histS[8][16 * HSTRIDE]; // per-wave tile buffer
  __shared__ __align__(16) bf16  tgtb[128];
  __shared__ float tgtf[128];
  __shared__ float bias0[80];       // t@(W0b - W0c) + b0
  __shared__ float scoresS[208];
  __shared__ float attnS[208];
  __shared__ float red[256];
  __shared__ float uiP[8][128];

  const int b    = blockIdx.x;
  const int tid  = threadIdx.x;
  const int wave = tid >> 5;
  const int lane = tid & 31;
  const int mrow = lane & 15;
  const int hi   = lane >> 4;

  if (tid < 128) {
    float v = tgt[(size_t)b*DIM + tid];
    tgtf[tid] = v;
    tgtb[tid] = (bf16)v;
  }
  __syncthreads();

  // per-batch effective bias for layer 0 (tgt contribution folded in, fp32)
  if (tid < 80) {
    float acc = b0[tid];
    #pragma unroll 4
    for (int k = 0; k < 128; ++k)
      acc += tgtf[k] * (W0[(128 + k)*80 + tid] - W0[(256 + k)*80 + tid]);
    bias0[tid] = acc;
  }
  __syncthreads();

  const bf16* W0ac = ws + WS_W0AC;
  const bf16* W0d  = ws + WS_W0D;
  const bf16* W1t  = ws + WS_W1;

  for (int t = wave; t < 13; t += 8) {          // 13 row-tiles of 16 (L padded to 208)
    bf16* hs = &histS[wave][0];

    // stage hist rows into LDS as bf16 (vectorized: b128 loads, b64 stores)
    #pragma unroll 4
    for (int j = lane; j < 512; j += 32) {      // 512 float4 chunks per 16x128 tile
      int r = j >> 5;                           // 32 float4 per row
      int c = (j & 31) * 4;
      int row = t*16 + r;
      v4f v = {0.f, 0.f, 0.f, 0.f};
      if (row < SEQ)
        v = *(const v4f*)(hist + ((size_t)b*SEQ + row)*DIM + c);
      v4bf bv;
      #pragma unroll
      for (int q = 0; q < 4; ++q) bv[q] = (bf16)v[q];
      *(v4bf*)(hs + r*HSTRIDE + c) = bv;
    }
    asm volatile("s_wait_dscnt 0x0" ::: "memory");
    __builtin_amdgcn_wave_barrier();

    // ---- layer 0: acc(16x80) = hist@W0ac + (hist*tgt)@W0d + bias0 ----
    v8f acc[NT0];
    #pragma unroll
    for (int nt = 0; nt < NT0; ++nt) {
      float bv = bias0[nt*16 + mrow];
      #pragma unroll
      for (int i = 0; i < 8; ++i) acc[nt][i] = bv;
    }
    #pragma unroll
    for (int kt = 0; kt < KT0; ++kt) {
      const int k0 = kt*32;
      const int c0 = k0 + hi*16;
      const int c1 = k0 + hi*16 + 8;
      const bf16* rowp = hs + mrow*HSTRIDE;
      v8bf hA = *(const v8bf*)(rowp + c0);
      v8bf hB = *(const v8bf*)(rowp + c1);
      v8bf tA = *(const v8bf*)(tgtb + c0);
      v8bf tB = *(const v8bf*)(tgtb + c1);
      v16bf a_h, a_p;
      #pragma unroll
      for (int i = 0; i < 8; ++i) {
        a_h[i]     = hA[i];
        a_h[8 + i] = hB[i];
        a_p[i]     = (bf16)((float)hA[i] * (float)tA[i]);   // prod on the fly
        a_p[8 + i] = (bf16)((float)hB[i] * (float)tB[i]);
      }
      #pragma unroll
      for (int nt = 0; nt < NT0; ++nt) {
        v16bf bac = *(const v16bf*)(W0ac + ((kt*NT0 + nt) << 10) + lane*16);
        v16bf bd  = *(const v16bf*)(W0d  + ((kt*NT0 + nt) << 10) + lane*16);
        acc[nt] = wmma_bf16(a_h, bac, acc[nt]);
        acc[nt] = wmma_bf16(a_p, bd,  acc[nt]);
      }
    }

    // relu -> store h0 tile (16 x 96 bf16) back into hs (reuse as scratch)
    asm volatile("s_wait_dscnt 0x0" ::: "memory");
    __builtin_amdgcn_wave_barrier();
    #pragma unroll
    for (int nt = 0; nt < NT0; ++nt) {
      #pragma unroll
      for (int r = 0; r < 8; ++r) {
        int row = hi ? (8 + r) : r;
        hs[row*H0STRIDE + nt*16 + mrow] = (bf16)fmaxf(acc[nt][r], 0.f);
      }
    }
    #pragma unroll
    for (int r = 0; r < 8; ++r) {       // zero K-padding cols 80..95
      int row = hi ? (8 + r) : r;
      hs[row*H0STRIDE + 80 + mrow] = (bf16)0.f;
    }
    asm volatile("s_wait_dscnt 0x0" ::: "memory");
    __builtin_amdgcn_wave_barrier();

    // ---- layer 1: h1(16x40) = relu(h0 @ W1 + b1) ----
    v8f acc2[NT1];
    #pragma unroll
    for (int nt = 0; nt < NT1; ++nt) {
      int n = nt*16 + mrow;
      float bv = (n < 40) ? b1[n] : 0.f;
      #pragma unroll
      for (int i = 0; i < 8; ++i) acc2[nt][i] = bv;
    }
    #pragma unroll
    for (int kt = 0; kt < KT1; ++kt) {
      const int k0 = kt*32;
      const int c0 = k0 + hi*16;
      const int c1 = k0 + hi*16 + 8;
      const bf16* rowp = hs + mrow*H0STRIDE;
      v8bf hA = *(const v8bf*)(rowp + c0);
      v8bf hB = *(const v8bf*)(rowp + c1);
      v16bf a;
      #pragma unroll
      for (int i = 0; i < 8; ++i) { a[i] = hA[i]; a[8 + i] = hB[i]; }
      #pragma unroll
      for (int nt = 0; nt < NT1; ++nt) {
        v16bf bw = *(const v16bf*)(W1t + ((kt*NT1 + nt) << 10) + lane*16);
        acc2[nt] = wmma_bf16(a, bw, acc2[nt]);
      }
    }

    // ---- layer 2: score = relu(h1) . W2 + b2 (cross-lane reduce) ----
    float w2v[NT1];
    #pragma unroll
    for (int nt = 0; nt < NT1; ++nt) {
      int n = nt*16 + mrow;
      w2v[nt] = (n < 40) ? W2[n] : 0.f;
    }
    float bias2 = b2[0];
    #pragma unroll
    for (int r = 0; r < 8; ++r) {
      float p = 0.f;
      #pragma unroll
      for (int nt = 0; nt < NT1; ++nt)
        p += fmaxf(acc2[nt][r], 0.f) * w2v[nt];
      p += __shfl_xor(p, 1, 32);
      p += __shfl_xor(p, 2, 32);
      p += __shfl_xor(p, 4, 32);
      p += __shfl_xor(p, 8, 32);   // sum within each 16-lane half
      if (mrow == 0) {
        int row = t*16 + (hi ? (8 + r) : r);
        scoresS[row] = p + bias2;
      }
    }
  }
  __syncthreads();

  // ---- masked softmax over L ----
  float sc = -3.0e38f;
  if (tid < SEQ) {
    float m = (float)mask[(size_t)b*SEQ + tid];
    sc = scoresS[tid] + (1.f - m) * -1.0e9f;
  }
  red[tid] = sc;
  __syncthreads();
  for (int off = 128; off > 0; off >>= 1) {
    if (tid < off) red[tid] = fmaxf(red[tid], red[tid + off]);
    __syncthreads();
  }
  float mx = red[0];
  __syncthreads();
  float e = (tid < SEQ) ? __expf(sc - mx) : 0.f;
  red[tid] = e;
  __syncthreads();
  for (int off = 128; off > 0; off >>= 1) {
    if (tid < off) red[tid] += red[tid + off];
    __syncthreads();
  }
  float inv = 1.f / red[0];
  float av = e * inv;
  if (tid < SEQ) {
    attnS[tid] = av;
    out_attn[(size_t)b*SEQ + tid] = av;
  }
  __syncthreads();

  // ---- user_interest = attn @ hist (fp32, vectorized b128, L2-resident) ----
  {
    int col = lane * 4;                 // each lane owns 4 consecutive dims
    float a0 = 0.f, a1 = 0.f, a2 = 0.f, a3 = 0.f;
    for (int l = wave; l < SEQ; l += 8) {   // each wave owns a row-slice
      float aw = attnS[l];
      v4f h = *(const v4f*)(hist + ((size_t)b*SEQ + l)*DIM + col);
      a0 += aw * h[0];
      a1 += aw * h[1];
      a2 += aw * h[2];
      a3 += aw * h[3];
    }
    uiP[wave][col + 0] = a0;
    uiP[wave][col + 1] = a1;
    uiP[wave][col + 2] = a2;
    uiP[wave][col + 3] = a3;
  }
  __syncthreads();
  if (tid < 128) {
    float s = 0.f;
    #pragma unroll
    for (int g = 0; g < 8; ++g) s += uiP[g][tid];
    out_ui[(size_t)b*DIM + tid] = s;
  }
}

extern "C" void kernel_launch(void* const* d_in, const int* in_sizes, int n_in,
                              void* d_out, int out_size, void* d_ws, size_t ws_size,
                              hipStream_t stream)
{
  const float* hist = (const float*)d_in[0];
  const float* tgt  = (const float*)d_in[1];
  const int*   mask = (const int*)  d_in[2];
  const float* W0   = (const float*)d_in[3];
  const float* b0   = (const float*)d_in[4];
  const float* W1   = (const float*)d_in[5];
  const float* b1   = (const float*)d_in[6];
  const float* W2   = (const float*)d_in[7];
  const float* b2   = (const float*)d_in[8];
  bf16* ws = (bf16*)d_ws;
  float* out_ui   = (float*)d_out;
  float* out_attn = out_ui + (size_t)BATCH * DIM;

  (void)in_sizes; (void)n_in; (void)out_size; (void)ws_size;

  din_prep<<<(WS_TOTAL + 255) / 256, 256, 0, stream>>>(W0, W1, ws);
  din_main<<<BATCH, 256, 0, stream>>>(hist, tgt, mask, W0, b0, b1, W2, b2,
                                      ws, out_ui, out_attn);
}
